// MultiHeadAttention_50809463112207
// MI455X (gfx1250) — matrix operand order repaired and checked by reading, not compiled, not run on hardware
//
#include <hip/hip_runtime.h>
#include <hip/hip_bf16.h>
#include <stdint.h>

typedef __bf16 bf16_t;
typedef __attribute__((ext_vector_type(16))) __bf16 v16bf;
typedef __attribute__((ext_vector_type(8)))  __bf16 v8bf;
typedef __attribute__((ext_vector_type(8)))  float  v8f;

constexpr int Ed = 1024;   // EMBED_DIM
constexpr int Hn = 16;     // HEADS
constexpr int DK = 64;     // d_k
constexpr int Bn = 2;      // batch
constexpr int Sn = 2048;   // seq len

union bfpair { uint32_t u; bf16_t h[2]; };

__device__ __forceinline__ v8f wmma_bf16(const v16bf& a, const v16bf& b, v8f c) {
  // D(f32 16x16) = A(bf16 16x32) * B(bf16 32x16) + C
  return __builtin_amdgcn_wmma_f32_16x16x32_bf16(false, a, false, b, (short)0, c,
                                                 false, false);
}

// WMMA 16-bit fragment addressing (ISA 7.12.2):
//   A: lane = row + 16*((k>>3)&1),  elem = ((k>>4)<<3) | (k&7)
//   B: lane = col + 16*((k>>3)&1), elem = ((k>>4)<<3) | (k&7)
//   C/D: vgpr r -> row r + 8*(lane>>4), col = lane&15

// ---------------------------------------------------------------------------
// Kernel 1: fused QKV projections.  Y = X @ W^T  (W stored [out,in]).
// grid = (Ed/64, (Bn*Sn)/64, 3), block = 128 (4 waves).
// Block tile 64M x 64N: wave wv owns N-subtile wv, sweeps 4 M-subtiles
// -> 16 WMMA per block per 32-k step, 4x B reuse.
// z=0: query @ Wq^T -> Q  [B,H,S,dk]   (bf16)
// z=1: key   @ Wk^T -> Kt [B,H,dk,S]   (bf16, transposed for QK^T B-frags)
// z=2: value @ Wv^T -> V  [B,H,S,dk]   (bf16)
// ---------------------------------------------------------------------------
__global__ __launch_bounds__(128) void qkv_proj(
    const float* __restrict__ q_in, const float* __restrict__ k_in,
    const float* __restrict__ v_in, const float* __restrict__ Wq,
    const float* __restrict__ Wk,   const float* __restrict__ Wv,
    bf16_t* __restrict__ Qo, bf16_t* __restrict__ Kto, bf16_t* __restrict__ Vo)
{
  // LDS tiles in WMMA fragment order -> b128 fragment reads
  __shared__ __align__(32) bf16_t As_lin[4][512];     // 4 M-subtiles, 16r x 32k
  __shared__ __align__(32) bf16_t Bs_lin[4][512];     // 4 N-subtiles, 32k x 16n

  const int z = blockIdx.z;
  const float* __restrict__ X = (z == 0) ? q_in : (z == 1) ? k_in : v_in;
  const float* __restrict__ W = (z == 0) ? Wq   : (z == 1) ? Wk   : Wv;

  const int mBase = blockIdx.y * 64;
  const int nBase = blockIdx.x * 64;
  const int tid   = threadIdx.x;
  const int lane  = tid & 31;
  const int wv    = tid >> 5;
  const int half  = lane >> 4;
  const int col   = lane & 15;

  v8f acc[4] = {v8f{}, v8f{}, v8f{}, v8f{}};

  for (int k0 = 0; k0 < Ed; k0 += 32) {
    // ---- stage A: 64 rows x 32 k = 1024 pairs, float2 loads, b32 stores ----
#pragma unroll
    for (int i = 0; i < 8; ++i) {
      int p  = tid + 128 * i;          // pair index 0..1023
      int r  = p >> 4;                 // row 0..63
      int kp = (p & 15) * 2;           // even k 0..30
      float2 xv = *(const float2*)&X[(mBase + r) * Ed + k0 + kp];
      int tl = (r & 15) + (((kp >> 3) & 1) << 4);
      int te = ((kp >> 4) << 3) | (kp & 7);          // even
      bfpair pr; pr.h[0] = (bf16_t)xv.x; pr.h[1] = (bf16_t)xv.y;
      *(uint32_t*)&As_lin[r >> 4][tl * 16 + te] = pr.u;
    }
    // ---- stage B: 1024 n-pairs (two W rows per thread), b16 stores ----
#pragma unroll
    for (int i = 0; i < 8; ++i) {
      int p = tid + 128 * i;
      int k = p & 31;
      int n = (p >> 5) * 2;            // even n 0..62
      float w0 = W[(nBase + n)     * Ed + k0 + k];
      float w1 = W[(nBase + n + 1) * Ed + k0 + k];
      int tl = (n & 15) + (((k >> 3) & 1) << 4);
      int te = ((k >> 4) << 3) | (k & 7);
      Bs_lin[n >> 4][tl * 16 + te] = (bf16_t)w0;
      Bs_lin[n >> 4][(tl + 1) * 16 + te] = (bf16_t)w1;
    }
    if (k0 + 32 < Ed)  // stream next weight slab toward L2 (global_prefetch_b8)
      __builtin_prefetch(&W[(nBase + (tid >> 1)) * Ed + k0 + 32 + (tid & 1) * 16], 0, 1);
    __syncthreads();

    v16bf b = *(const v16bf*)&Bs_lin[wv][lane * 16];   // 2x ds_load_b128
#pragma unroll
    for (int mi = 0; mi < 4; ++mi) {
      v16bf a = *(const v16bf*)&As_lin[mi][lane * 16];
      acc[mi] = wmma_bf16(a, b, acc[mi]);
    }
    __syncthreads();
  }

#pragma unroll
  for (int mi = 0; mi < 4; ++mi)
#pragma unroll
    for (int r = 0; r < 8; ++r) {
      int m  = mBase + mi * 16 + r + 8 * half;
      int n  = nBase + wv * 16 + col;
      int bb = m / Sn, s = m % Sn;
      int h  = n / DK, dk = n % DK;
      bf16_t val = (bf16_t)acc[mi][r];
      if (z == 0)      Qo [((bb * Hn + h) * Sn + s)  * DK + dk] = val;
      else if (z == 1) Kto[((bb * Hn + h) * DK + dk) * Sn + s ] = val;
      else             Vo [((bb * Hn + h) * Sn + s)  * DK + dk] = val;
    }
}

// ---------------------------------------------------------------------------
// Kernel 2: causal flash attention, one wave per (b*h, 16 query rows).
// grid = (Sn/16, Bn*Hn), block = 32.  ctx output: [B,S,E] bf16
// ---------------------------------------------------------------------------
__global__ __launch_bounds__(32) void attn(
    const bf16_t* __restrict__ Q, const bf16_t* __restrict__ Kt,
    const bf16_t* __restrict__ V, bf16_t* __restrict__ ctx)
{
  __shared__ __align__(32) bf16_t ptile[512];  // P in A-fragment order

  const int q0   = blockIdx.x * 16;
  const int bh   = blockIdx.y;
  const int lane = threadIdx.x;
  const int half = lane >> 4;
  const int col  = lane & 15;

  const bf16_t* __restrict__ Qb  = Q  + (size_t)bh * Sn * DK;
  const bf16_t* __restrict__ Ktb = Kt + (size_t)bh * DK * Sn;
  const bf16_t* __restrict__ Vb  = V  + (size_t)bh * Sn * DK;

  // Q A-fragments (two dk halves): two aligned 16B chunks each
  v16bf qf[2];
#pragma unroll
  for (int hh = 0; hh < 2; ++hh) {
    const bf16_t* qb = &Qb[(q0 + col) * DK + 32 * hh + half * 8];
    v8bf c0 = *(const v8bf*)qb;          // k = base .. base+7   -> e 0..7
    v8bf c1 = *(const v8bf*)(qb + 16);   // k = base+16..base+23 -> e 8..15
    qf[hh] = __builtin_shufflevector(c0, c1, 0, 1, 2, 3, 4, 5, 6, 7,
                                     8, 9, 10, 11, 12, 13, 14, 15);
  }

  v8f acc[4] = {v8f{}, v8f{}, v8f{}, v8f{}};   // 16 x 64 f32 context
  float mst[8], lst[8];
#pragma unroll
  for (int r = 0; r < 8; ++r) { mst[r] = -1e30f; lst[r] = 0.0f; }

  const int nkb = (q0 + 15) / 32 + 1;
  for (int kb = 0; kb < nkb; ++kb) {
    const int kpBase = kb * 32;

    // ---- scores: two 16x16 subtiles over 32 key positions ----
    v8f sf[2];
#pragma unroll
    for (int j = 0; j < 2; ++j) {
      const int kp0 = kpBase + 16 * j;
      // B-frags: lane = key position col, elements = dk in table order (gathered)
      v16bf kf0, kf1;
#pragma unroll
      for (int s = 0; s < 16; ++s) {
        int kk = (s < 8) ? (8 * half + s) : (16 + 8 * half + (s - 8));
        kf0[s] = Ktb[(kk)      * Sn + kp0 + col];
        kf1[s] = Ktb[(32 + kk) * Sn + kp0 + col];
      }
      v8f c = {};
      c = wmma_bf16(qf[0], kf0, c);
      c = wmma_bf16(qf[1], kf1, c);
#pragma unroll
      for (int r = 0; r < 8; ++r) {
        int qrow = q0 + r + 8 * half;
        int kpos = kp0 + col;
        float sv = c[r] * 0.125f;                 // 1/sqrt(64)
        sf[j][r] = (kpos <= qrow) ? sv : -1e30f;  // causal mask
      }
    }

    // ---- online softmax: row reductions across 16 lanes of each half ----
#pragma unroll
    for (int r = 0; r < 8; ++r) {
      float mx = fmaxf(sf[0][r], sf[1][r]);
#pragma unroll
      for (int m = 8; m >= 1; m >>= 1) mx = fmaxf(mx, __shfl_xor(mx, m, 16));
      float mnew = fmaxf(mst[r], mx);
      float scl  = __expf(mst[r] - mnew);
      mst[r] = mnew;
      sf[0][r] = __expf(sf[0][r] - mnew);
      sf[1][r] = __expf(sf[1][r] - mnew);
      float sum = sf[0][r] + sf[1][r];
#pragma unroll
      for (int m = 8; m >= 1; m >>= 1) sum += __shfl_xor(sum, m, 16);
      lst[r] = lst[r] * scl + sum;
#pragma unroll
      for (int t = 0; t < 4; ++t) acc[t][r] *= scl;
    }

    // ---- P: C-layout -> A-fragment order in LDS (scatter write, b128 read) --
    __syncthreads();
#pragma unroll
    for (int j = 0; j < 2; ++j)
#pragma unroll
      for (int r = 0; r < 8; ++r) {
        int row  = r + 8 * half;
        int kcol = 16 * j + col;
        int tl = row + (((kcol >> 3) & 1) << 4);
        int te = ((kcol >> 4) << 3) | (kcol & 7);
        ptile[tl * 16 + te] = (bf16_t)sf[j][r];
      }
    __syncthreads();
    v16bf pf = *(const v16bf*)&ptile[lane * 16];

    // ---- PV: 4 n-tiles of 16 dk columns, gathered V loads (table k order) ----
#pragma unroll
    for (int t = 0; t < 4; ++t) {
      v16bf vf;
#pragma unroll
      for (int s = 0; s < 16; ++s) {
        int kk = (s < 8) ? (8 * half + s) : (16 + 8 * half + (s - 8));
        vf[s] = Vb[(kpBase + kk) * DK + 16 * t + col];
      }
      acc[t] = wmma_bf16(pf, vf, acc[t]);
    }
  }

  // ---- epilogue: normalize and store ctx [B,S,E] bf16 ----
  const int bb = bh / Hn, h = bh % Hn;
#pragma unroll
  for (int t = 0; t < 4; ++t)
#pragma unroll
    for (int r = 0; r < 8; ++r) {
      int qrow = q0 + r + 8 * half;
      int dk   = 16 * t + col;
      ctx[(size_t)(bb * Sn + qrow) * Ed + h * DK + dk] =
          (bf16_t)(acc[t][r] / lst[r]);
    }
}

// ---------------------------------------------------------------------------
// Kernel 3: output projection  out = ctx @ Wo^T + bo   (f32 output)
// grid = (Ed/64, (Bn*Sn)/64), block = 128.  Same 64x64 tiling as kernel 1.
// ---------------------------------------------------------------------------
__global__ __launch_bounds__(128) void out_proj(
    const bf16_t* __restrict__ ctx, const float* __restrict__ Wo,
    const float* __restrict__ bo, float* __restrict__ out)
{
  __shared__ __align__(32) bf16_t As_lin[4][512];
  __shared__ __align__(32) bf16_t Bs_lin[4][512];

  const int mBase = blockIdx.y * 64;
  const int nBase = blockIdx.x * 64;
  const int tid   = threadIdx.x;
  const int lane  = tid & 31;
  const int wv    = tid >> 5;
  const int half  = lane >> 4;
  const int col   = lane & 15;

  v8f acc[4] = {v8f{}, v8f{}, v8f{}, v8f{}};

  for (int k0 = 0; k0 < Ed; k0 += 32) {
    // A tile already bf16: copy pairs as raw u32
#pragma unroll
    for (int i = 0; i < 8; ++i) {
      int p  = tid + 128 * i;
      int r  = p >> 4;
      int kp = (p & 15) * 2;
      uint32_t v = *(const uint32_t*)&ctx[(size_t)(mBase + r) * Ed + k0 + kp];
      int tl = (r & 15) + (((kp >> 3) & 1) << 4);
      int te = ((kp >> 4) << 3) | (kp & 7);
      *(uint32_t*)&As_lin[r >> 4][tl * 16 + te] = v;
    }
#pragma unroll
    for (int i = 0; i < 8; ++i) {
      int p = tid + 128 * i;
      int k = p & 31;
      int n = (p >> 5) * 2;
      float w0 = Wo[(nBase + n)     * Ed + k0 + k];
      float w1 = Wo[(nBase + n + 1) * Ed + k0 + k];
      int tl = (n & 15) + (((k >> 3) & 1) << 4);
      int te = ((k >> 4) << 3) | (k & 7);
      Bs_lin[n >> 4][tl * 16 + te] = (bf16_t)w0;
      Bs_lin[n >> 4][(tl + 1) * 16 + te] = (bf16_t)w1;
    }
    if (k0 + 32 < Ed)
      __builtin_prefetch(&Wo[(nBase + (tid >> 1)) * Ed + k0 + 32 + (tid & 1) * 16], 0, 1);
    __syncthreads();

    v16bf b = *(const v16bf*)&Bs_lin[wv][lane * 16];
#pragma unroll
    for (int mi = 0; mi < 4; ++mi) {
      v16bf a = *(const v16bf*)&As_lin[mi][lane * 16];
      acc[mi] = wmma_bf16(a, b, acc[mi]);
    }
    __syncthreads();
  }

#pragma unroll
  for (int mi = 0; mi < 4; ++mi)
#pragma unroll
    for (int r = 0; r < 8; ++r) {
      int m = mBase + mi * 16 + r + 8 * half;
      int n = nBase + wv * 16 + col;
      out[(size_t)m * Ed + n] = acc[mi][r] + bo[n];
    }
}

// ---------------------------------------------------------------------------
// Launch: inputs in setup_inputs() order:
// 0:key 1:query 2:value 3:Wq 4:Wk 5:Wv 6:Wo 7:bo 8:mask(causal, ignored)
// Workspace: Q(8MB) + Kt(8MB) + V(8MB) + ctx(8MB) bf16 = 32MB
// ---------------------------------------------------------------------------
extern "C" void kernel_launch(void* const* d_in, const int* in_sizes, int n_in,
                              void* d_out, int out_size, void* d_ws, size_t ws_size,
                              hipStream_t stream) {
  const float* key_in   = (const float*)d_in[0];
  const float* query_in = (const float*)d_in[1];
  const float* value_in = (const float*)d_in[2];
  const float* Wq = (const float*)d_in[3];
  const float* Wk = (const float*)d_in[4];
  const float* Wv = (const float*)d_in[5];
  const float* Wo = (const float*)d_in[6];
  const float* bo = (const float*)d_in[7];

  const size_t elems = (size_t)Bn * Sn * Ed;  // 4,194,304
  bf16_t* Qw  = (bf16_t*)d_ws;
  bf16_t* Ktw = Qw  + elems;
  bf16_t* Vw  = Ktw + elems;
  bf16_t* Cw  = Vw  + elems;

  qkv_proj<<<dim3(Ed / 64, (Bn * Sn) / 64, 3), dim3(128), 0, stream>>>(
      query_in, key_in, value_in, Wq, Wk, Wv, Qw, Ktw, Vw);

  attn<<<dim3(Sn / 16, Bn * Hn), dim3(32), 0, stream>>>(Qw, Ktw, Vw, Cw);

  out_proj<<<dim3(Ed / 64, (Bn * Sn) / 64), dim3(128), 0, stream>>>(
      Cw, Wo, bo, (float*)d_out);
}
